// DCNV2_Ref_40596030882007
// MI455X (gfx1250) — compile-verified, matrix operand order checked by reading, not run on hardware
//
#include <hip/hip_runtime.h>
#include <hip/hip_bf16.h>
#include <math.h>

// ---------------------------------------------------------------------------
// DCNv2 forward for MI455X (gfx1250): bf16 WMMA GEMMs + NHWC bilinear gather
// + async global->LDS double-buffered staging for the offset conv.
// B=4, C=64, H=W=128, Cout=64, K=3x3, stride=1, pad=1, dil=1, G=1.
// ---------------------------------------------------------------------------

typedef __bf16 bf16;
typedef __attribute__((ext_vector_type(8)))  bf16  v8bf;
typedef __attribute__((ext_vector_type(16))) bf16  v16bf;
typedef __attribute__((ext_vector_type(8)))  float v8f;

static __device__ __forceinline__ v16bf catbf(v8bf lo, v8bf hi) {
  return __builtin_shufflevector(lo, hi, 0,1,2,3,4,5,6,7,8,9,10,11,12,13,14,15);
}

// ----------------------------- workspace map -------------------------------
// xT   : NHWC f32 x                  [4][128][128][64]       16,777,216 B
// cat  : NHWC bf16 concat(x,ref)     [4][130][130][128]      17,305,600 B  (1px zero halo)
// dy/dx/mk : f32                     [4][9][128][128] each    2,359,296 B
// Aoff : bf16 weights stage-1        [32][1152]                  73,728 B
// Adef : bf16 weights stage-2        [64][576]                   73,728 B
#define WS_XT    0
#define WS_CAT   16777216
#define WS_DY    34082816
#define WS_DX    36442112
#define WS_MK    38801408
#define WS_AOFF  41160704
#define WS_ADEF  41234432
#define CAT_U4   1081600   // 17,305,600 / 16

// --------------------------- kernel Z: zero cat ----------------------------
__global__ __launch_bounds__(256) void k_zero(uint4* __restrict__ p, int n) {
  int i = blockIdx.x * 256 + threadIdx.x;
  uint4 z = {0u, 0u, 0u, 0u};
  if (i < n) p[i] = z;
}

// --------------------------- kernel 0: pack --------------------------------
// NCHW -> NHWC via 32x32 LDS tile transpose; also emits bf16 concat(x,ref)
// into the haloed cat buffer (interior only; halo stays zero).
__global__ __launch_bounds__(256) void k_pack(const float* __restrict__ x,
                                              const float* __restrict__ ref,
                                              float* __restrict__ xT,
                                              bf16*  __restrict__ cat) {
  __shared__ float tx[32][33];
  __shared__ float tr[32][33];
  int bi = blockIdx.x;
  int c0 = (bi & 1) * 32;
  int w0 = ((bi >> 1) & 3) * 32;
  int h  = (bi >> 3) & 127;
  int b  = bi >> 10;
  int tid = threadIdx.x;
  int wi = tid & 31;
  int cj = tid >> 5;
  for (int ci = cj; ci < 32; ci += 8) {
    int g = ((b * 64 + c0 + ci) * 128 + h) * 128 + w0 + wi;
    tx[ci][wi] = x[g];
    tr[ci][wi] = ref[g];
  }
  __syncthreads();
  int ci2 = tid & 31;
  int wj  = tid >> 5;
  for (int wi2 = wj; wi2 < 32; wi2 += 8) {
    int pix = (b * 128 + h) * 128 + w0 + wi2;
    int cpix = (b * 130 + (h + 1)) * 130 + (w0 + wi2 + 1);
    float vx = tx[ci2][wi2];
    float vr = tr[ci2][wi2];
    xT[pix * 64 + c0 + ci2]         = vx;
    cat[cpix * 128 + c0 + ci2]      = (bf16)vx;
    cat[cpix * 128 + 64 + c0 + ci2] = (bf16)vr;
  }
}

// ------------------------ kernel W: weight repack --------------------------
__global__ __launch_bounds__(256) void k_weights(const float* __restrict__ w_offset,
                                                 const float* __restrict__ w_deform,
                                                 bf16* __restrict__ Aoff,
                                                 bf16* __restrict__ Adef) {
  int idx = blockIdx.x * 256 + threadIdx.x;
  if (idx < 32 * 1152) {
    int m = idx / 1152, kk = idx % 1152;
    int t = kk >> 7, c = kk & 127;
    float v = 0.f;
    if (m < 27) v = w_offset[((m * 128 + c) * 3 + t / 3) * 3 + (t % 3)];
    Aoff[idx] = (bf16)v;
  } else if (idx < 32 * 1152 + 64 * 576) {
    int i2 = idx - 32 * 1152;
    int m = i2 / 576, kk = i2 % 576;
    int t = kk >> 6, c = kk & 63;
    Adef[i2] = (bf16)w_deform[((m * 64 + c) * 3 + t / 3) * 3 + (t % 3)];
  }
}

// -------------------- async staging helper (stage 1) -----------------------
// Streams one tap chunk [64 px][128 ch] bf16 from the haloed cat buffer
// directly into LDS; exactly 4 async-load instructions per wave, EXEC all-1s.
static __device__ __forceinline__ void fill_async(const bf16* __restrict__ cat,
                                                  bf16* dstBase,
                                                  int b, int oh, int ow0, int t,
                                                  int tid) {
  int ih1 = oh + t / 3;        // halo row:   (oh + t/3 - 1) + 1
  int dwp = t % 3;             // halo shift: (t%3 - 1) + 1
#pragma unroll
  for (int j = 0; j < 4; ++j) {
    int jj = tid + j * 256;
    int p = jj >> 4, q = jj & 15;
    int iw1 = ow0 + p + dwp;
    unsigned long long src =
        (unsigned long long)(cat + (((b * 130 + ih1) * 130 + iw1) * 128 + q * 8));
    unsigned lds = (unsigned)(unsigned long long)(dstBase + p * 136 + q * 8);
    asm volatile("global_load_async_to_lds_b128 %0, %1, off"
                 :: "v"(lds), "v"(src) : "memory");
  }
}

// ---------------------- kernel 1: offset conv (WMMA) -----------------------
// GEMM M=32(pad 27), K=1152, N=65536. Block: 64 px row-segment, 8 waves,
// each wave one 16x16 tile. K chunked per tap (128); LDS double-buffered
// with async global->LDS copies overlapped against the WMMAs.
__global__ __launch_bounds__(256) void k_offset_conv(const bf16* __restrict__ cat,
                                                     const bf16* __restrict__ Aoff,
                                                     const float* __restrict__ b_offset,
                                                     float* __restrict__ dy,
                                                     float* __restrict__ dx,
                                                     float* __restrict__ mk) {
  __shared__ __align__(16) bf16 patch[2][64][136];
  int tid = threadIdx.x, lane = tid & 31, wv = tid >> 5;
  int pb = blockIdx.x * 64;
  int b = pb >> 14, oh = (pb >> 7) & 127, ow0 = pb & 127;
  int mt = wv >> 2, nt = wv & 3;
  int mrow  = mt * 16 + (lane & 15);
  int khalf = (lane & 16) ? 8 : 0;
  int prow  = nt * 16 + (lane & 15);
  v8f acc = {0.f, 0.f, 0.f, 0.f, 0.f, 0.f, 0.f, 0.f};

  fill_async(cat, &patch[0][0][0], b, oh, ow0, 0, tid);

  for (int t = 0; t < 9; ++t) {
    int cur = t & 1;
    if (t < 8) {
      fill_async(cat, &patch[cur ^ 1][0][0], b, oh, ow0, t + 1, tid);
      asm volatile("s_wait_asynccnt 0x4" ::: "memory");  // my fill(t) landed
    } else {
      asm volatile("s_wait_asynccnt 0x0" ::: "memory");
    }
    __syncthreads();  // everyone's fill(t) landed
    const bf16* arow = Aoff + mrow * 1152 + t * 128;
#pragma unroll
    for (int ks = 0; ks < 4; ++ks) {
      int kc = ks * 32 + khalf;
      v16bf a  = catbf(*(const v8bf*)(arow + kc),
                       *(const v8bf*)(arow + kc + 16));
      v16bf bb = catbf(*(const v8bf*)(&patch[cur][prow][kc]),
                       *(const v8bf*)(&patch[cur][prow][kc + 16]));
      acc = __builtin_amdgcn_wmma_f32_16x16x32_bf16(false, a, false, bb,
                                                    (short)0, acc, false, false);
    }
    __syncthreads();  // all reads of patch[cur] done -> refillable next iter
  }

#pragma unroll
  for (int r = 0; r < 8; ++r) {
    int m = mt * 16 + r + ((lane & 16) ? 8 : 0);
    int n = nt * 16 + (lane & 15);
    int sp = oh * 128 + (ow0 + n);
    if (m < 27) {
      float v = acc[r] + b_offset[m];
      if (m < 9)       dy[(b * 9 + m)        * 16384 + sp] = v;
      else if (m < 18) dx[(b * 9 + (m - 9))  * 16384 + sp] = v;
      else             mk[(b * 9 + (m - 18)) * 16384 + sp] = 1.f / (1.f + expf(-v));
    }
  }
}

// --------------------- kernel 2: deformable conv (WMMA) --------------------
__global__ __launch_bounds__(256) void k_deform(const float* __restrict__ xT,
                                                const bf16*  __restrict__ Adef,
                                                const float* __restrict__ dy,
                                                const float* __restrict__ dx,
                                                const float* __restrict__ mk,
                                                const float* __restrict__ b_deform,
                                                float* __restrict__ out) {
  __shared__ __align__(16) bf16 patch[32][584];
  int tid = threadIdx.x, lane = tid & 31, wv = tid >> 5;
  int pb = blockIdx.x * 32;
  int b = pb >> 14, oh = (pb >> 7) & 127, ow0 = pb & 127;

  // Phase A: 32 px * 9 taps * 8 channel-groups = 2304 jobs, 9 per thread.
  for (int j = tid; j < 2304; j += 256) {
    int p = j / 72;
    int rem = j - p * 72;
    int t = rem >> 3, cg = rem & 7;
    int ow = ow0 + p;
    int oidx = (b * 9 + t) * 16384 + oh * 128 + ow;
    float dyv = dy[oidx], dxv = dx[oidx], mv = mk[oidx];
    float py = (float)(oh + t / 3 - 1) + dyv;
    float px = (float)(ow + t % 3 - 1) + dxv;
    float y0f = floorf(py), x0f = floorf(px);
    float wy = py - y0f, wx = px - x0f;
    int y0 = (int)y0f, x0 = (int)x0f;
    int y1 = y0 + 1, x1 = x0 + 1;
    float w00 = (1.f - wy) * (1.f - wx), w01 = (1.f - wy) * wx;
    float w10 = wy * (1.f - wx),         w11 = wy * wx;
    if ((unsigned)y0 >= 128u) { w00 = 0.f; w01 = 0.f; }
    if ((unsigned)y1 >= 128u) { w10 = 0.f; w11 = 0.f; }
    if ((unsigned)x0 >= 128u) { w00 = 0.f; w10 = 0.f; }
    if ((unsigned)x1 >= 128u) { w01 = 0.f; w11 = 0.f; }
    int yc0 = min(max(y0, 0), 127), yc1 = min(max(y1, 0), 127);
    int xc0 = min(max(x0, 0), 127), xc1 = min(max(x1, 0), 127);
    int c8 = cg * 8;
    v8f q00 = *(const v8f*)(xT + (((b * 128 + yc0) * 128 + xc0) * 64 + c8));
    v8f q01 = *(const v8f*)(xT + (((b * 128 + yc0) * 128 + xc1) * 64 + c8));
    v8f q10 = *(const v8f*)(xT + (((b * 128 + yc1) * 128 + xc0) * 64 + c8));
    v8f q11 = *(const v8f*)(xT + (((b * 128 + yc1) * 128 + xc1) * 64 + c8));
    v8bf o;
#pragma unroll
    for (int i = 0; i < 8; ++i)
      o[i] = (bf16)((w00 * q00[i] + w01 * q01[i] + w10 * q10[i] + w11 * q11[i]) * mv);
    *(v8bf*)(&patch[p][t * 64 + c8]) = o;
  }
  __syncthreads();

  // Phase B: GEMM M=64, K=576, N=32; wave tile (mt in 0..3, nt in 0..1).
  int mt = wv >> 1, nt = wv & 1;
  int mrow  = mt * 16 + (lane & 15);
  int khalf = (lane & 16) ? 8 : 0;
  int prow  = nt * 16 + (lane & 15);
  v8f acc = {0.f, 0.f, 0.f, 0.f, 0.f, 0.f, 0.f, 0.f};
  const bf16* arow = Adef + mrow * 576;
  for (int ks = 0; ks < 18; ++ks) {
    int kc = ks * 32 + khalf;
    v16bf a  = catbf(*(const v8bf*)(arow + kc),
                     *(const v8bf*)(arow + kc + 16));
    v16bf bb = catbf(*(const v8bf*)(&patch[prow][kc]),
                     *(const v8bf*)(&patch[prow][kc + 16]));
    acc = __builtin_amdgcn_wmma_f32_16x16x32_bf16(false, a, false, bb,
                                                  (short)0, acc, false, false);
  }

#pragma unroll
  for (int r = 0; r < 8; ++r) {
    int cout = mt * 16 + r + ((lane & 16) ? 8 : 0);
    int n = nt * 16 + (lane & 15);
    out[((b * 64 + cout) * 128 + oh) * 128 + (ow0 + n)] = acc[r] + b_deform[cout];
  }
}

// ------------------------------- launcher ----------------------------------
extern "C" void kernel_launch(void* const* d_in, const int* in_sizes, int n_in,
                              void* d_out, int out_size, void* d_ws, size_t ws_size,
                              hipStream_t stream) {
  (void)in_sizes; (void)n_in; (void)out_size; (void)ws_size;
  const float* x        = (const float*)d_in[0];
  const float* ref      = (const float*)d_in[1];
  const float* w_offset = (const float*)d_in[2];
  const float* b_offset = (const float*)d_in[3];
  const float* w_deform = (const float*)d_in[4];
  const float* b_deform = (const float*)d_in[5];
  char* ws = (char*)d_ws;
  float* xT   = (float*)(ws + WS_XT);
  bf16*  cat  = (bf16*)(ws + WS_CAT);
  float* dy   = (float*)(ws + WS_DY);
  float* dx   = (float*)(ws + WS_DX);
  float* mk   = (float*)(ws + WS_MK);
  bf16*  Aoff = (bf16*)(ws + WS_AOFF);
  bf16*  Adef = (bf16*)(ws + WS_ADEF);
  float* out  = (float*)d_out;

  hipLaunchKernelGGL(k_zero,        dim3(4225), dim3(256), 0, stream, (uint4*)cat, CAT_U4);
  hipLaunchKernelGGL(k_pack,        dim3(4096), dim3(256), 0, stream, x, ref, xT, cat);
  hipLaunchKernelGGL(k_weights,     dim3(288),  dim3(256), 0, stream, w_offset, w_deform, Aoff, Adef);
  hipLaunchKernelGGL(k_offset_conv, dim3(1024), dim3(256), 0, stream, cat, Aoff, b_offset, dy, dx, mk);
  hipLaunchKernelGGL(k_deform,      dim3(2048), dim3(256), 0, stream, xT, Adef, dy, dx, mk, b_deform, out);
}
